// ARPredVideoVanilla_48395691491375
// MI455X (gfx1250) — matrix operand order repaired and checked by reading, not compiled, not run on hardware
//
#include <hip/hip_runtime.h>
#include <hip/hip_bf16.h>
#include <math.h>

// ---------------------------------------------------------------------------
// Model dimensions (compile-time constants from the reference)
// ---------------------------------------------------------------------------
#define BB   8
#define TT   4
#define CC   12
#define PP   128          // patches per frame
#define SS   512          // T * P
#define DM   768          // encoder dim
#define NH   12
#define HD   64
#define LL   8
#define DDm  512          // decoder dim
#define NHD  8
#define LDd  4
#define MLPe 3072
#define MLPd 2048
#define OUTd 768          // PH*PW*V

typedef __attribute__((ext_vector_type(16))) __bf16 v16bf;
typedef __attribute__((ext_vector_type(8)))  __bf16 v8bf;
typedef __attribute__((ext_vector_type(4)))  __bf16 v4bf;
typedef __attribute__((ext_vector_type(8)))  float  v8f;
typedef __attribute__((ext_vector_type(4)))  float  v4f;

// ---------------------------------------------------------------------------
// Generic WMMA GEMM:  C = alpha * A @ B (+bias) (gelu) (+residual)
//   A: fp32 [M x K] row-major (lda).  B: fp32; TRANSB ? B[k][n]=Bp[n*ldb+k]
//   : B[k][n]=Bp[k*ldb+n].  Batched via blockIdx.z = outer*inner with dual
//   strides.  ALL problem dims are exact multiples of the tile dims, so
//   staging and epilogue are guard-free.
//
//   Software-pipelined: global loads for tile i+1 are issued right after the
//   LDS of tile i is populated, so they overlap the fragment loads + WMMAs
//   (single s_wait_loadcnt per tile, a full compute stage after issue).
//   fp32 -> bf16 conversion happens in registers (v_cvt_pk_bf16_f32) between
//   the b128 global loads and b64 LDS stores; math is
//   v_wmma_f32_16x16x32_bf16 with fp32 accumulation.
//   LDS operand layouts are fragment-contiguous:
//     Alds[m][k]  rows padded to 40 bf16 (80 B)  -> A frag = 2x ds_load_b128
//     Blds[n][k]  (transposed) same padding      -> B frag = 2x ds_load_b128
// ---------------------------------------------------------------------------
template<int BM, int BN, int BK, int WGM, int WGN, int WMT, int WNT, int TRANSB>
__global__ __launch_bounds__(256)
void gemm_wmma_kernel(const float* __restrict__ A, const float* __restrict__ B,
                      float* __restrict__ C,
                      const float* __restrict__ bias,
                      const float* __restrict__ res,
                      int M, int N, int K, int lda, int ldb, int ldc,
                      long long sAb, long long sAh,
                      long long sBb, long long sBh,
                      long long sCb, long long sCh,
                      int innerB, float alpha, int doGelu)
{
    static_assert(WGM * WGN == 8, "8 waves / 256 threads");
    static_assert(WGM * WMT * 16 == BM, "BM tiling");
    static_assert(WGN * WNT * 16 == BN, "BN tiling");
    static_assert(BK == 32, "one WMMA K-step per tile");

    constexpr int LDK = BK + 8;          // 40 bf16 = 80 B row pitch (16B-aligned,
                                         // 16 consecutive rows hit distinct banks)
    __shared__ __bf16 Alds[BM][LDK];
    __shared__ __bf16 Blds[BN][LDK];     // B stored transposed: [n][k]

    const int z  = blockIdx.z;
    const int zb = z / innerB;
    const int zh = z % innerB;
    const float* Ab = A + (long long)zb * sAb + (long long)zh * sAh;
    const float* Bb = B + (long long)zb * sBb + (long long)zh * sBh;
    float*       Cb = C + (long long)zb * sCb + (long long)zh * sCh;
    const float* Rb = res ? (res + (long long)zb * sCb + (long long)zh * sCh) : nullptr;

    const int tid  = threadIdx.x;
    const int lane = tid & 31;
    const int w    = tid >> 5;
    const int half = lane >> 4;     // lane group (selects K/M halves per ISA layout)
    const int lr   = lane & 15;

    const int wm = (w % WGM) * (WMT * 16);
    const int wn = (w / WGM) * (WNT * 16);
    const int m0 = blockIdx.y * BM;
    const int n0 = blockIdx.x * BN;

    v8f acc[WMT][WNT];
#pragma unroll
    for (int mt = 0; mt < WMT; ++mt)
#pragma unroll
        for (int nt = 0; nt < WNT; ++nt)
#pragma unroll
            for (int v = 0; v < 8; ++v) acc[mt][nt][v] = 0.0f;

    // staging decomposition: quads of 4 k-values, fixed (m|n, kq) per thread
    constexpr int AQT = BM * (BK / 4) / 256;   // A float4 quads per thread
    constexpr int BQT = BN * (BK / 4) / 256;   // B quads per thread

    int am[AQT], akq[AQT];
#pragma unroll
    for (int i = 0; i < AQT; ++i) {
        int q = i * 256 + tid;
        am[i]  = q >> 3;           // / (BK/4)
        akq[i] = q & 7;            // % (BK/4)
    }
    int bn[BQT], bkq[BQT];
#pragma unroll
    for (int i = 0; i < BQT; ++i) {
        int q = i * 256 + tid;
        bn[i]  = q % BN;           // consecutive lanes -> consecutive n
        bkq[i] = q / BN;
    }

    v4f regA[AQT], regB[BQT];

    // ---- prologue: load tile 0 into registers ----
#pragma unroll
    for (int i = 0; i < AQT; ++i)
        regA[i] = *(const v4f*)(Ab + (long long)(m0 + am[i]) * lda + akq[i] * 4);
#pragma unroll
    for (int i = 0; i < BQT; ++i) {
        if (TRANSB) {
            regB[i] = *(const v4f*)(Bb + (long long)(n0 + bn[i]) * ldb + bkq[i] * 4);
        } else {
#pragma unroll
            for (int j = 0; j < 4; ++j)
                regB[i][j] = Bb[(long long)(bkq[i] * 4 + j) * ldb + (n0 + bn[i])];
        }
    }

    for (int k0 = 0; k0 < K; k0 += BK) {
        // ---- cvt + store current tile registers to LDS ----
#pragma unroll
        for (int i = 0; i < AQT; ++i) {
            v4bf pk;
            pk[0] = (__bf16)regA[i][0]; pk[1] = (__bf16)regA[i][1];
            pk[2] = (__bf16)regA[i][2]; pk[3] = (__bf16)regA[i][3];
            *(v4bf*)(&Alds[am[i]][akq[i] * 4]) = pk;
        }
#pragma unroll
        for (int i = 0; i < BQT; ++i) {
            v4bf pk;
            pk[0] = (__bf16)regB[i][0]; pk[1] = (__bf16)regB[i][1];
            pk[2] = (__bf16)regB[i][2]; pk[3] = (__bf16)regB[i][3];
            *(v4bf*)(&Blds[bn[i]][bkq[i] * 4]) = pk;
        }
        __syncthreads();

        // ---- prefetch next tile (overlaps fragment loads + WMMAs below) ----
        const int k1 = k0 + BK;
        if (k1 < K) {
#pragma unroll
            for (int i = 0; i < AQT; ++i)
                regA[i] = *(const v4f*)(Ab + (long long)(m0 + am[i]) * lda + (k1 + akq[i] * 4));
#pragma unroll
            for (int i = 0; i < BQT; ++i) {
                if (TRANSB) {
                    regB[i] = *(const v4f*)(Bb + (long long)(n0 + bn[i]) * ldb + (k1 + bkq[i] * 4));
                } else {
#pragma unroll
                    for (int j = 0; j < 4; ++j)
                        regB[i][j] = Bb[(long long)(k1 + bkq[i] * 4 + j) * ldb + (n0 + bn[i])];
                }
            }
        }

        // ---- fragments: each = two 16B contiguous LDS loads ----
        v16bf afrag[WMT];
        v16bf bfrag[WNT];
#pragma unroll
        for (int mt = 0; mt < WMT; ++mt) {
            const __bf16* ap = &Alds[wm + mt * 16 + lr][0];
            v8bf lo = *(const v8bf*)(ap + half * 8);        // K = half*8 .. +7
            v8bf hi = *(const v8bf*)(ap + 16 + half * 8);   // K = 16+half*8 .. +7
            afrag[mt] = __builtin_shufflevector(lo, hi, 0,1,2,3,4,5,6,7,
                                                        8,9,10,11,12,13,14,15);
        }
#pragma unroll
        for (int nt = 0; nt < WNT; ++nt) {
            const __bf16* bp = &Blds[wn + nt * 16 + lr][0];
            v8bf lo = *(const v8bf*)(bp + half * 8);        // K = half*8 .. +7
            v8bf hi = *(const v8bf*)(bp + 16 + half * 8);   // K = 16+half*8 .. +7
            bfrag[nt] = __builtin_shufflevector(lo, hi, 0,1,2,3,4,5,6,7,
                                                        8,9,10,11,12,13,14,15);
        }

        // ---- WMT*WNT WMMAs per wave per K-step (8 for 128x128, 4 for 128x64) ----
#pragma unroll
        for (int mt = 0; mt < WMT; ++mt)
#pragma unroll
            for (int nt = 0; nt < WNT; ++nt)
                acc[mt][nt] = __builtin_amdgcn_wmma_f32_16x16x32_bf16(
                    false, afrag[mt], false, bfrag[nt],
                    (short)0, acc[mt][nt], false, false);
        __syncthreads();
    }

    // ---- epilogue: alpha, bias, gelu, residual (C layout: M = v + 8*half) ----
#pragma unroll
    for (int mt = 0; mt < WMT; ++mt) {
#pragma unroll
        for (int nt = 0; nt < WNT; ++nt) {
            const int col = n0 + wn + nt * 16 + lr;
            const float bv = bias ? bias[col] : 0.0f;
#pragma unroll
            for (int v = 0; v < 8; ++v) {
                const int row = m0 + wm + mt * 16 + half * 8 + v;
                float val = alpha * acc[mt][nt][v] + bv;
                if (doGelu) val = 0.5f * val * (1.0f + erff(val * 0.70710678118654752f));
                if (Rb) val += Rb[(long long)row * ldc + col];
                Cb[(long long)row * ldc + col] = val;
            }
        }
    }
}

// ---------------------------------------------------------------------------
// Elementwise / normalization / softmax kernels (fp32)
// ---------------------------------------------------------------------------
__global__ void patch_gather_kernel(const float* __restrict__ x, float* __restrict__ xf)
{
    long long i = (long long)blockIdx.x * 256 + threadIdx.x;
    const long long n = (long long)4096 * 3072;
    if (i >= n) return;
    int col  = (int)(i % 3072);
    int rowp = (int)(i / 3072);     // = bt*128 + p
    int bt = rowp >> 7;
    int p  = rowp & 127;
    int hp = p >> 4, wp = p & 15;
    int c  = col >> 8;
    int ph = (col >> 4) & 15, pw = col & 15;
    int h = hp * 16 + ph, wcol = wp * 16 + pw;
    xf[i] = x[((long long)bt * CC + c) * 32768 + h * 256 + wcol];
}

__global__ void add_pos_kernel(float* __restrict__ t, const float* __restrict__ pos,
                               int n, int per)
{
    int i = blockIdx.x * 256 + threadIdx.x;
    if (i < n) t[i] += pos[i % per];
}

__global__ void bcast_kernel(float* __restrict__ dst, const float* __restrict__ src,
                             int n, int per)
{
    int i = blockIdx.x * 256 + threadIdx.x;
    if (i < n) dst[i] = src[i % per];
}

__global__ void layernorm_kernel(const float* __restrict__ in, float* __restrict__ out,
                                 const float* __restrict__ s, const float* __restrict__ b,
                                 int width)
{
    const int row = blockIdx.x;
    const float* r = in + (long long)row * width;
    float* o = out + (long long)row * width;
    __shared__ float red[256];
    float sum = 0.0f, sq = 0.0f;
    for (int i = threadIdx.x; i < width; i += 256) {
        float v = r[i]; sum += v; sq += v * v;
    }
    red[threadIdx.x] = sum; __syncthreads();
    for (int st = 128; st > 0; st >>= 1) {
        if (threadIdx.x < st) red[threadIdx.x] += red[threadIdx.x + st];
        __syncthreads();
    }
    float mean = red[0] / width; __syncthreads();
    red[threadIdx.x] = sq; __syncthreads();
    for (int st = 128; st > 0; st >>= 1) {
        if (threadIdx.x < st) red[threadIdx.x] += red[threadIdx.x + st];
        __syncthreads();
    }
    float var = red[0] / width - mean * mean;
    float inv = rsqrtf(var + 1e-5f);
    for (int i = threadIdx.x; i < width; i += 256)
        o[i] = (r[i] - mean) * inv * s[i] + b[i];
}

// One block per attention row. Mask derived from mask_rand block structure:
// block (tq,tk) masked iff tk>tq && mask_rand[l,tq,tk] < 0.8  (P=128 blocks).
__global__ void softmax_kernel(float* __restrict__ att, int S, int rowsPerBatch,
                               const float* __restrict__ mask_rand, int layer)
{
    const int row = blockIdx.x;
    const int q = row % rowsPerBatch;
    const int tq = q >> 7;
    float* r = att + (long long)row * S;
    __shared__ float red[256];

    float mx = -INFINITY;
    for (int k = threadIdx.x; k < S; k += 256) {
        float v = r[k];
        if (mask_rand) {
            int tk = k >> 7;
            if (tk > tq && mask_rand[layer * (TT * TT) + tq * TT + tk] < 0.8f) {
                v = -INFINITY;
                r[k] = v;
            }
        }
        mx = fmaxf(mx, v);
    }
    red[threadIdx.x] = mx; __syncthreads();
    for (int st = 128; st > 0; st >>= 1) {
        if (threadIdx.x < st) red[threadIdx.x] = fmaxf(red[threadIdx.x], red[threadIdx.x + st]);
        __syncthreads();
    }
    mx = red[0]; __syncthreads();

    float sum = 0.0f;
    for (int k = threadIdx.x; k < S; k += 256) {
        float e = __expf(r[k] - mx);
        r[k] = e;
        sum += e;
    }
    red[threadIdx.x] = sum; __syncthreads();
    for (int st = 128; st > 0; st >>= 1) {
        if (threadIdx.x < st) red[threadIdx.x] += red[threadIdx.x + st];
        __syncthreads();
    }
    float inv = 1.0f / red[0];
    for (int k = threadIdx.x; k < S; k += 256) r[k] *= inv;
}

// ---------------------------------------------------------------------------
// Host-side launch helpers
// ---------------------------------------------------------------------------
static inline void gemm128x128(hipStream_t st,
    const float* A, const float* B, float* C, const float* bias, const float* res,
    int M, int N, int K, int lda, int ldb, int ldc,
    long long sAb, long long sAh, long long sBb, long long sBh,
    long long sCb, long long sCh,
    int outerB, int innerB, float alpha, int transB, int gelu)
{
    dim3 g((N + 127) / 128, (M + 127) / 128, outerB * innerB);
    if (transB)
        gemm_wmma_kernel<128, 128, 32, 4, 2, 2, 4, 1><<<g, 256, 0, st>>>(
            A, B, C, bias, res, M, N, K, lda, ldb, ldc,
            sAb, sAh, sBb, sBh, sCb, sCh, innerB, alpha, gelu);
    else
        gemm_wmma_kernel<128, 128, 32, 4, 2, 2, 4, 0><<<g, 256, 0, st>>>(
            A, B, C, bias, res, M, N, K, lda, ldb, ldc,
            sAb, sAh, sBb, sBh, sCb, sCh, innerB, alpha, gelu);
}

static inline void gemm128x64(hipStream_t st,
    const float* A, const float* B, float* C, const float* bias, const float* res,
    int M, int N, int K, int lda, int ldb, int ldc,
    long long sAb, long long sAh, long long sBb, long long sBh,
    long long sCb, long long sCh,
    int outerB, int innerB, float alpha, int gelu)
{
    // only ever used with transB = 0 (P @ V)
    dim3 g((N + 63) / 64, (M + 127) / 128, outerB * innerB);
    gemm_wmma_kernel<128, 64, 32, 4, 2, 2, 2, 0><<<g, 256, 0, st>>>(
        A, B, C, bias, res, M, N, K, lda, ldb, ldc,
        sAb, sAh, sBb, sBh, sCb, sCh, innerB, alpha, gelu);
}

static inline void gemm_plain(hipStream_t st,
    const float* A, const float* B, float* C, const float* bias, const float* res,
    int M, int N, int K, int ldb_, int transB, int gelu)
{
    gemm128x128(st, A, B, C, bias, res, M, N, K,
                K, ldb_, N, 0, 0, 0, 0, 0, 0, 1, 1, 1.0f, transB, gelu);
}

// ---------------------------------------------------------------------------
// Orchestration
// ---------------------------------------------------------------------------
extern "C" void kernel_launch(void* const* d_in, const int* in_sizes, int n_in,
                              void* d_out, int out_size, void* d_ws, size_t ws_size,
                              hipStream_t stream)
{
    (void)in_sizes; (void)n_in; (void)out_size; (void)ws_size;

    const float* x          = (const float*)d_in[0];
    const float* mask_rand  = (const float*)d_in[1];
    const float* conv_w     = (const float*)d_in[2];
    const float* conv_b     = (const float*)d_in[3];
    const float* pos_emb    = (const float*)d_in[4];
    const float* enc_ln1_s  = (const float*)d_in[5];
    const float* enc_ln1_b  = (const float*)d_in[6];
    const float* enc_qkv_w  = (const float*)d_in[7];
    const float* enc_proj_w = (const float*)d_in[8];
    const float* enc_proj_b = (const float*)d_in[9];
    const float* enc_ln2_s  = (const float*)d_in[10];
    const float* enc_ln2_b  = (const float*)d_in[11];
    const float* enc_mlp_w1 = (const float*)d_in[12];
    const float* enc_mlp_b1 = (const float*)d_in[13];
    const float* enc_mlp_w2 = (const float*)d_in[14];
    const float* enc_mlp_b2 = (const float*)d_in[15];
    const float* e2d_w      = (const float*)d_in[16];
    const float* e2d_b      = (const float*)d_in[17];
    const float* dec_query  = (const float*)d_in[18];
    const float* dec_ln1_s  = (const float*)d_in[19];
    const float* dec_ln1_b  = (const float*)d_in[20];
    const float* dec_qkv_w  = (const float*)d_in[21];
    const float* dec_qkv_b  = (const float*)d_in[22];
    const float* dec_out_w  = (const float*)d_in[23];
    const float* dec_out_b  = (const float*)d_in[24];
    const float* dec_ln2_s  = (const float*)d_in[25];
    const float* dec_ln2_b  = (const float*)d_in[26];
    const float* dec_mlp_w1 = (const float*)d_in[27];
    const float* dec_mlp_b1 = (const float*)d_in[28];
    const float* dec_mlp_w2 = (const float*)d_in[29];
    const float* dec_mlp_b2 = (const float*)d_in[30];
    const float* head_ln_s  = (const float*)d_in[31];
    const float* head_ln_b  = (const float*)d_in[32];
    const float* head_w     = (const float*)d_in[33];
    const float* head_b     = (const float*)d_in[34];

    float* out = (float*)d_out;
    float* ws  = (float*)d_ws;

    // ---- workspace arenas (fp32 elements), unioned across phases ----
    const long long N_M  = 4096;                  // B*S (encoder rows)
    const long long N_Md = 1024;                  // B*P (decoder rows)

    float* xf    = ws;                            // 4096*3072 = 12,582,912
    float* mlpu  = ws;                            // reuse of xf arena
    float* arenB = ws + 12582912LL;               // 25,165,824 (enc attention)
    float* att   = arenB;                         // 96*512*512
    float* attd  = arenB;                         // 64*128*512 (decoder, reuse)
    float* mem   = arenB + 20971520LL;            // 4096*512 (lives in arenB tail)
    float* qkv   = ws + 12582912LL + 25165824LL;  // 4096*2304 = 9,437,184
    float* Km    = qkv;                           // 4096*512 (decoder reuse)
    float* Vm    = qkv + 2097152LL;               // 4096*512
    float* Qm    = qkv + 4194304LL;               // 1024*512
    float* h     = qkv + 9437184LL;               // 4096*768 = 3,145,728
    float* qd    = h;                             // 1024*512 (decoder reuse)
    float* y     = h + 3145728LL;                 // 4096*768
    float* o     = y + 3145728LL;                 // 4096*768
    float* od    = o;                             // 1024*512
    float* ud    = o + 524288LL;                  // 1024*2048

    const float scale = 0.125f;   // HD^-0.5 = HDD^-0.5 = 1/8

    // ---- 1. patchify: gather xf, tok = xf @ conv_w.T + conv_b (+pos) ----
    {
        long long n = N_M * 3072;
        patch_gather_kernel<<<(unsigned)((n + 255) / 256), 256, 0, stream>>>(x, xf);
        gemm_plain(stream, xf, conv_w, h, conv_b, nullptr,
                   (int)N_M, DM, 3072, /*ldb*/3072, /*transB*/1, 0);
        int np = (int)(N_M * DM);
        add_pos_kernel<<<(np + 255) / 256, 256, 0, stream>>>(h, pos_emb, np, SS * DM);
    }

    // ---- 2. encoder layers ----
    for (int l = 0; l < LL; ++l) {
        layernorm_kernel<<<(unsigned)N_M, 256, 0, stream>>>(
            h, y, enc_ln1_s + l * DM, enc_ln1_b + l * DM, DM);

        // qkv = y @ W_qkv  (no bias in encoder)
        gemm_plain(stream, y, enc_qkv_w + (long long)l * DM * 3 * DM, qkv,
                   nullptr, nullptr, (int)N_M, 3 * DM, DM, 3 * DM, 0, 0);

        // scores = scale * Q @ K^T   (batch: outer=B, inner=NH)
        gemm128x128(stream, qkv, qkv + DM, att, nullptr, nullptr,
                    SS, SS, HD, 3 * DM, 3 * DM, SS,
                    (long long)SS * 3 * DM, HD,
                    (long long)SS * 3 * DM, HD,
                    (long long)NH * SS * SS, (long long)SS * SS,
                    BB, NH, scale, /*transB*/1, 0);

        softmax_kernel<<<BB * NH * SS, 256, 0, stream>>>(att, SS, SS, mask_rand, l);

        // O = P @ V   (N = 64)
        gemm128x64(stream, att, qkv + 2 * DM, o, nullptr, nullptr,
                   SS, HD, SS, SS, 3 * DM, DM,
                   (long long)NH * SS * SS, (long long)SS * SS,
                   (long long)SS * 3 * DM, HD,
                   (long long)SS * DM, HD,
                   BB, NH, 1.0f, 0);

        // h = h + O @ W_proj + b_proj
        gemm_plain(stream, o, enc_proj_w + (long long)l * DM * DM, h,
                   enc_proj_b + l * DM, h, (int)N_M, DM, DM, DM, 0, 0);

        layernorm_kernel<<<(unsigned)N_M, 256, 0, stream>>>(
            h, y, enc_ln2_s + l * DM, enc_ln2_b + l * DM, DM);

        // u = gelu(y @ W1 + b1);  h = h + u @ W2 + b2
        gemm_plain(stream, y, enc_mlp_w1 + (long long)l * DM * MLPe, mlpu,
                   enc_mlp_b1 + l * MLPe, nullptr, (int)N_M, MLPe, DM, MLPe, 0, 1);
        gemm_plain(stream, mlpu, enc_mlp_w2 + (long long)l * MLPe * DM, h,
                   enc_mlp_b2 + l * DM, h, (int)N_M, DM, MLPe, DM, 0, 0);
    }

    // ---- 3. encoder -> decoder memory ----
    gemm_plain(stream, h, e2d_w, mem, e2d_b, nullptr, (int)N_M, DDm, DM, DDm, 0, 0);

    // ---- 4. decoder query broadcast (overwrites h arena -> qd) ----
    {
        int nq = (int)(N_Md * DDm);
        bcast_kernel<<<(nq + 255) / 256, 256, 0, stream>>>(qd, dec_query, nq, PP * DDm);
    }

    // ---- 5. decoder layers (cross attention) ----
    for (int l = 0; l < LDd; ++l) {
        layernorm_kernel<<<(unsigned)N_Md, 256, 0, stream>>>(
            qd, y, dec_ln1_s + l * DDm, dec_ln1_b + l * DDm, DDm);

        const float* Wq = dec_qkv_w + (long long)(l * 3 + 0) * DDm * DDm;
        const float* Wk = dec_qkv_w + (long long)(l * 3 + 1) * DDm * DDm;
        const float* Wv = dec_qkv_w + (long long)(l * 3 + 2) * DDm * DDm;
        const float* bq = dec_qkv_b + (l * 3 + 0) * DDm;
        const float* bk = dec_qkv_b + (l * 3 + 1) * DDm;
        const float* bv = dec_qkv_b + (l * 3 + 2) * DDm;

        gemm_plain(stream, y,   Wq, Qm, bq, nullptr, (int)N_Md, DDm, DDm, DDm, 0, 0);
        gemm_plain(stream, mem, Wk, Km, bk, nullptr, (int)N_M,  DDm, DDm, DDm, 0, 0);
        gemm_plain(stream, mem, Wv, Vm, bv, nullptr, (int)N_M,  DDm, DDm, DDm, 0, 0);

        // scores = dscale * Q @ K^T  (outer=B, inner=NHD)
        gemm128x128(stream, Qm, Km, attd, nullptr, nullptr,
                    PP, SS, HD, DDm, DDm, SS,
                    (long long)PP * DDm, HD,
                    (long long)SS * DDm, HD,
                    (long long)NHD * PP * SS, (long long)PP * SS,
                    BB, NHD, scale, /*transB*/1, 0);

        softmax_kernel<<<BB * NHD * PP, 256, 0, stream>>>(attd, SS, PP, nullptr, 0);

        // O = P @ V
        gemm128x64(stream, attd, Vm, od, nullptr, nullptr,
                   PP, HD, SS, SS, DDm, DDm,
                   (long long)NHD * PP * SS, (long long)PP * SS,
                   (long long)SS * DDm, HD,
                   (long long)PP * DDm, HD,
                   BB, NHD, 1.0f, 0);

        // q = q + O @ W_out + b_out
        gemm_plain(stream, od, dec_out_w + (long long)l * DDm * DDm, qd,
                   dec_out_b + l * DDm, qd, (int)N_Md, DDm, DDm, DDm, 0, 0);

        layernorm_kernel<<<(unsigned)N_Md, 256, 0, stream>>>(
            qd, y, dec_ln2_s + l * DDm, dec_ln2_b + l * DDm, DDm);

        gemm_plain(stream, y, dec_mlp_w1 + (long long)l * DDm * MLPd, ud,
                   dec_mlp_b1 + l * MLPd, nullptr, (int)N_Md, MLPd, DDm, MLPd, 0, 1);
        gemm_plain(stream, ud, dec_mlp_w2 + (long long)l * MLPd * DDm, qd,
                   dec_mlp_b2 + l * DDm, qd, (int)N_Md, DDm, MLPd, DDm, 0, 0);
    }

    // ---- 6. head: out = LN(q) @ head_w + head_b ----
    layernorm_kernel<<<(unsigned)N_Md, 256, 0, stream>>>(qd, y, head_ln_s, head_ln_b, DDm);
    gemm_plain(stream, y, head_w, out, head_b, nullptr, (int)N_Md, OUTd, DDm, OUTd, 0, 0);
}
